// Net_47596827574272
// MI455X (gfx1250) — matrix operand order repaired and checked by reading, not compiled, not run on hardware
//
#include <hip/hip_runtime.h>
#include <math.h>
#include <stdint.h>

typedef __attribute__((ext_vector_type(16))) _Float16 v16h;
typedef __attribute__((ext_vector_type(4)))  _Float16 v4h;
typedef __attribute__((ext_vector_type(8)))  float    v8f;
typedef __attribute__((ext_vector_type(4)))  float    v4f;

#define GRAPHS 256
#define NODES  64
#define KNN    24
#define HID    128

__device__ __forceinline__ float elu_f(float x) {
    return x > 0.0f ? x : expm1f(x);
}

// ---------------------------------------------------------------------------
// Kernel 1: weight prep (f16, transposed so WMMA B-fragments are contiguous
// row reads).
//   W1T[n][k] (128x32): W1[k][n] zero-padded K 15->32   (layer-1 WMMA)
//   W2T[n][k] (128x128)                                  (layer-2 WMMA)
//   WcT[n][k] (256x128): cols 0..127 = Wc_top - Wc_bot, 128..255 = Wc_bot
// ---------------------------------------------------------------------------
__global__ void prep_weights(const float* __restrict__ W1,
                             const float* __restrict__ W2,
                             const float* __restrict__ Wc,
                             _Float16* __restrict__ W1T,
                             _Float16* __restrict__ W2T,
                             _Float16* __restrict__ WcT) {
    int tid = blockIdx.x * blockDim.x + threadIdx.x;
    if (tid < 128 * 32) {
        int n = tid / 32, k = tid % 32;
        W1T[tid] = (k < 15) ? (_Float16)W1[k * 128 + n] : (_Float16)0.0f;
    } else if (tid < 128 * 32 + 128 * 128) {
        int t2 = tid - 128 * 32;
        int n = t2 / 128, k = t2 % 128;
        W2T[t2] = (_Float16)W2[k * 128 + n];
    } else if (tid < 128 * 32 + 128 * 128 + 256 * 128) {
        int t2 = tid - 128 * 32 - 128 * 128;
        int n = t2 / 128, k = t2 % 128;
        float v;
        if (n < 128) v = Wc[k * 128 + n] - Wc[(128 + k) * 128 + n];
        else         v = Wc[(128 + k) * 128 + (n - 128)];
        WcT[t2] = (_Float16)v;
    }
}

// ---------------------------------------------------------------------------
// Kernel 2: input MLP, both layers through v_wmma_f32_16x16x32_f16.
// Layer 1: K padded 15->32 (zero halves), one K-step per tile.
// ---------------------------------------------------------------------------
__global__ void mlp_in(const float* __restrict__ x,      // 16384 x 15
                       const _Float16* __restrict__ W1T, // 128 x 32 (T, padded)
                       const float* __restrict__ b1,
                       const _Float16* __restrict__ W2T, // 128 x 128 (T)
                       const float* __restrict__ b2,
                       float* __restrict__ h) {          // 16384 x 128
    __shared__ _Float16 xs1[64 * 32];   // padded input tile
    __shared__ _Float16 t1[64 * 128];   // layer-1 activations
    const int row0 = blockIdx.x * 64;
    const int tid  = threadIdx.x;

    // stage padded input tile
    for (int p = tid; p < 64 * 32; p += 256) xs1[p] = (_Float16)0.0f;
    __syncthreads();
    for (int p = tid; p < 64 * 15; p += 256) {
        int r = p / 15, c = p % 15;
        xs1[r * 32 + c] = (_Float16)x[(size_t)(row0 + r) * 15 + c];
    }
    __syncthreads();

    const int wave = tid >> 5, lane = tid & 31;
    const int t  = (lane >= 16) ? 1 : 0;
    const int ln = lane & 15;

    // layer 1: 32 tiles (4 M x 8 N), single K-step (K=32)
    for (int tile = wave; tile < 32; tile += 8) {
        int m0 = (tile >> 3) * 16, n0 = (tile & 7) * 16;
        v16h a, b;
        const _Float16* pa = &xs1[(m0 + ln) * 32 + 16 * t];
        const _Float16* pb = &W1T[(n0 + ln) * 32 + 16 * t];
        #pragma unroll
        for (int j = 0; j < 16; ++j) a[j] = pa[j];
        #pragma unroll
        for (int j = 0; j < 16; ++j) b[j] = pb[j];
        v8f acc = {};
        acc = __builtin_amdgcn_wmma_f32_16x16x32_f16(
            false, a, false, b, (short)0, acc, false, false);
        int n = n0 + ln;
        float bias = b1[n];
        #pragma unroll
        for (int r = 0; r < 8; ++r)
            t1[(m0 + r + 8 * t) * 128 + n] = (_Float16)elu_f(acc[r] + bias);
    }
    __syncthreads();

    // layer 2: 32 tiles (4 M x 8 N), 4 K-steps
    for (int tile = wave; tile < 32; tile += 8) {
        int m0 = (tile >> 3) * 16, n0 = (tile & 7) * 16;
        v8f acc = {};
        #pragma unroll
        for (int kt = 0; kt < 4; ++kt) {
            int k0 = kt * 32;
            v16h a, b;
            const _Float16* pa = &t1[(m0 + ln) * 128 + k0 + 16 * t];
            const _Float16* pb = &W2T[(n0 + ln) * 128 + k0 + 16 * t];
            #pragma unroll
            for (int j = 0; j < 16; ++j) a[j] = pa[j];
            #pragma unroll
            for (int j = 0; j < 16; ++j) b[j] = pb[j];
            acc = __builtin_amdgcn_wmma_f32_16x16x32_f16(
                false, a, false, b, (short)0, acc, false, false);
        }
        int n = n0 + ln;
        float bias = b2[n];
        #pragma unroll
        for (int r = 0; r < 8; ++r) {
            int m = m0 + r + 8 * t;
            h[(size_t)(row0 + m) * 128 + n] = elu_f(acc[r] + bias);
        }
    }
}

// ---------------------------------------------------------------------------
// Kernel 3: EdgeConv layer.  One block per graph (64 nodes x 128 feat).
//   tile load : GLOBAL_LOAD_ASYNC_TO_LDS_B128 (ASYNCcnt) into LDS staging
//   Gram      = xs @ xs^T            (WMMA; distances + norms from it)
//   AB        = xs @ [Wt-Wb|Wb]      (WMMA, 64x256)
//   out[i]    = elu( A_i + bc + max_k B_{knn(i,k)} )  (elu hoisted: monotone)
// LDS arena: xs(16K f16) | 32K region = f32 staging -> gram/dist -> ABs f16 |
//            nidx(64x24 u8) | sq(64 f32)  = ~51 KB.
// ---------------------------------------------------------------------------
__global__ void edge_conv(const float* __restrict__ xin,    // 16384 x 128
                          const _Float16* __restrict__ WcT, // 256 x 128 (T)
                          const float* __restrict__ bc,
                          float* __restrict__ xout) {       // 16384 x 128
    __shared__ __align__(16) unsigned char smem[16 * 1024 + 32 * 1024 + NODES * KNN + NODES * 4];
    _Float16* xs   = (_Float16*)smem;                 // 64 x 128 f16
    float*    stage= (float*)   (smem + 16 * 1024);   // 64 x 128 f32 (phase 1)
    _Float16* ABs  = (_Float16*)(smem + 16 * 1024);   // 64 x 256 f16 (phase 6+)
    float*    dmat = (float*)   (smem + 16 * 1024);   // 64 x 64 f32 (phase 2-5)
    unsigned char* nidx = smem + 48 * 1024;           // 64 x 24
    float*    sq   = (float*)(smem + 48 * 1024 + NODES * KNN);

    const int g   = blockIdx.x;
    const int tid = threadIdx.x;
    const float* xg = xin + (size_t)g * NODES * HID;

    // phase 1: async DMA f32 tile -> LDS staging, then convert to f16
    for (int p = tid; p < NODES * HID / 4; p += 256) {
        unsigned lds_addr = (unsigned)(uintptr_t)(stage + p * 4);
        const float* gp = xg + p * 4;
        asm volatile("global_load_async_to_lds_b128 %0, %1, off"
                     :: "v"(lds_addr), "v"(gp)
                     : "memory");
    }
    asm volatile("s_wait_asynccnt 0x0" ::: "memory");
    __syncthreads();
    for (int p = tid; p < NODES * HID / 4; p += 256) {
        float4 v = ((const float4*)stage)[p];
        v4h h4;
        h4[0] = (_Float16)v.x; h4[1] = (_Float16)v.y;
        h4[2] = (_Float16)v.z; h4[3] = (_Float16)v.w;
        *(v4h*)(xs + p * 4) = h4;
    }
    __syncthreads();

    const int wave = tid >> 5, lane = tid & 31;
    const int t  = (lane >= 16) ? 1 : 0;
    const int ln = lane & 15;

    // phase 2: Gram = xs @ xs^T via WMMA (B = xs^T fragments are contiguous
    // rows of xs).  16 tiles.
    for (int tile = wave; tile < 16; tile += 8) {
        int m0 = (tile >> 2) * 16, n0 = (tile & 3) * 16;
        v8f acc = {};
        #pragma unroll
        for (int kt = 0; kt < 4; ++kt) {
            int k0 = kt * 32;
            v16h a, b;
            const _Float16* pa = &xs[(m0 + ln) * HID + k0 + 16 * t];
            const _Float16* pb = &xs[(n0 + ln) * HID + k0 + 16 * t];
            #pragma unroll
            for (int j = 0; j < 16; ++j) a[j] = pa[j];
            #pragma unroll
            for (int j = 0; j < 16; ++j) b[j] = pb[j];
            acc = __builtin_amdgcn_wmma_f32_16x16x32_f16(
                false, a, false, b, (short)0, acc, false, false);
        }
        #pragma unroll
        for (int r = 0; r < 8; ++r)
            dmat[(m0 + r + 8 * t) * 64 + (n0 + ln)] = acc[r];
    }
    __syncthreads();

    // phase 3: norms from Gram diagonal
    if (tid < NODES) sq[tid] = dmat[tid * 64 + tid];
    __syncthreads();

    // phase 4: in-place distance transform  d = sq_i + sq_j - 2*g_ij
    for (int p = tid; p < NODES * NODES; p += 256) {
        int i = p >> 6, j = p & 63;
        dmat[p] = sq[i] + sq[j] - 2.0f * dmat[p];
    }
    __syncthreads();

    // phase 5: top-KNN smallest per row (first occurrence wins -> top_k(-d))
    if (tid < NODES) {
        float* drow = dmat + tid * NODES;
        for (int kk = 0; kk < KNN; ++kk) {
            float best = drow[0]; int bi = 0;
            for (int j = 1; j < NODES; ++j) {
                float v = drow[j];
                if (v < best) { best = v; bi = j; }
            }
            nidx[tid * KNN + kk] = (unsigned char)bi;
            drow[bi] = 3.0e38f;
        }
    }
    __syncthreads();   // dmat dead; ABs region may be overwritten now

    // phase 6: AB(64x256) = xs(64x128) @ WcT^T, WMMA f16 -> f32 acc
    for (int tile = wave; tile < 64; tile += 8) {
        int m0 = (tile >> 4) * 16, n0 = (tile & 15) * 16;
        v8f acc = {};
        #pragma unroll
        for (int kt = 0; kt < 4; ++kt) {
            int k0 = kt * 32;
            v16h a, b;
            const _Float16* pa = &xs[(m0 + ln) * HID + k0 + 16 * t];
            const _Float16* pb = &WcT[(n0 + ln) * HID + k0 + 16 * t];
            #pragma unroll
            for (int j = 0; j < 16; ++j) a[j] = pa[j];
            #pragma unroll
            for (int j = 0; j < 16; ++j) b[j] = pb[j];
            acc = __builtin_amdgcn_wmma_f32_16x16x32_f16(
                false, a, false, b, (short)0, acc, false, false);
        }
        #pragma unroll
        for (int r = 0; r < 8; ++r)
            ABs[(m0 + r + 8 * t) * 256 + (n0 + ln)] = (_Float16)acc[r];
    }
    __syncthreads();

    // phase 7: aggregate 4 channels/thread.
    // out[i,c] = elu( A_i[c] + bc[c] + max_k B_{idx(i,k)}[c] )
    for (int p = tid; p < NODES * HID / 4; p += 256) {
        int i  = p >> 5;
        int c0 = (p & 31) * 4;
        v4h av = *(const v4h*)(ABs + i * 256 + c0);
        float4 bcv = *(const float4*)(bc + c0);
        float a0 = (float)av[0] + bcv.x;
        float a1 = (float)av[1] + bcv.y;
        float a2 = (float)av[2] + bcv.z;
        float a3 = (float)av[3] + bcv.w;
        float m0v = -3.0e38f, m1v = -3.0e38f, m2v = -3.0e38f, m3v = -3.0e38f;
        const unsigned char* row = nidx + i * KNN;
        #pragma unroll 4
        for (int kk = 0; kk < KNN; ++kk) {
            int j = row[kk];
            v4h bv = *(const v4h*)(ABs + j * 256 + 128 + c0);   // ds_load_b64
            m0v = fmaxf(m0v, (float)bv[0]);
            m1v = fmaxf(m1v, (float)bv[1]);
            m2v = fmaxf(m2v, (float)bv[2]);
            m3v = fmaxf(m3v, (float)bv[3]);
        }
        v4f o;
        o[0] = elu_f(a0 + m0v);
        o[1] = elu_f(a1 + m1v);
        o[2] = elu_f(a2 + m2v);
        o[3] = elu_f(a3 + m3v);
        *(v4f*)(xout + (size_t)(g * NODES + i) * HID + c0) = o;   // b128 store
    }
}

// ---------------------------------------------------------------------------
// Kernel 4: pooling + MLP head + batch_pf tuple element.  One block per graph.
// ---------------------------------------------------------------------------
__global__ void head(const float* __restrict__ f3,
                     const float* __restrict__ Wo1, const float* __restrict__ bo1,
                     const float* __restrict__ Wo2, const float* __restrict__ bo2,
                     const float* __restrict__ Wo3, const float* __restrict__ bo3,
                     const float* __restrict__ Wo4, const float* __restrict__ bo4,
                     float* __restrict__ out) {
    __shared__ float pooled[128];
    __shared__ float o1[64];
    __shared__ float o2[32];
    __shared__ float o3[32];
    const int g = blockIdx.x, tid = threadIdx.x;

    if (tid < 128) {
        float s = 0.f;
        for (int i = 0; i < 64; ++i)
            s += f3[(size_t)(g * 64 + i) * 128 + tid];
        pooled[tid] = s;
    }
    __syncthreads();
    if (tid < 64) {
        float s = bo1[tid];
        for (int k = 0; k < 128; ++k) s = fmaf(pooled[k], Wo1[k * 64 + tid], s);
        o1[tid] = elu_f(s);
    }
    __syncthreads();
    if (tid < 32) {
        float s = bo2[tid];
        for (int k = 0; k < 64; ++k) s = fmaf(o1[k], Wo2[k * 32 + tid], s);
        o2[tid] = elu_f(s);
    }
    __syncthreads();
    if (tid < 32) {
        float s = bo3[tid];
        for (int k = 0; k < 32; ++k) s = fmaf(o2[k], Wo3[k * 32 + tid], s);
        o3[tid] = elu_f(s);
    }
    __syncthreads();
    if (tid < 8) {
        float s = bo4[tid];
        for (int k = 0; k < 32; ++k) s = fmaf(o3[k], Wo4[k * 8 + tid], s);
        out[g * 8 + tid] = s;
    }
    // second tuple output: batch_pf (int32) appended after 256*8 floats
    if (tid < 64) {
        int* ip = (int*)(out + GRAPHS * 8);
        ip[g * 64 + tid] = g;
    }
}

// ---------------------------------------------------------------------------
extern "C" void kernel_launch(void* const* d_in, const int* in_sizes, int n_in,
                              void* d_out, int out_size, void* d_ws, size_t ws_size,
                              hipStream_t stream) {
    const float* x_pf = (const float*)d_in[0];
    // d_in[1] = batch_pf (regenerated on device in head kernel)
    const float* W1  = (const float*)d_in[2];
    const float* b1  = (const float*)d_in[3];
    const float* W2  = (const float*)d_in[4];
    const float* b2  = (const float*)d_in[5];
    const float* Wc  = (const float*)d_in[6];
    const float* bc  = (const float*)d_in[7];
    const float* Wo1 = (const float*)d_in[8];
    const float* bo1 = (const float*)d_in[9];
    const float* Wo2 = (const float*)d_in[10];
    const float* bo2 = (const float*)d_in[11];
    const float* Wo3 = (const float*)d_in[12];
    const float* bo3 = (const float*)d_in[13];
    const float* Wo4 = (const float*)d_in[14];
    const float* bo4 = (const float*)d_in[15];

    char* ws = (char*)d_ws;
    _Float16* W1T = (_Float16*)ws;                        // 8 KB
    _Float16* W2T = (_Float16*)(ws + 8192);               // 32 KB
    _Float16* WcT = (_Float16*)(ws + 8192 + 32768);       // 64 KB
    float* bufA   = (float*)(ws + 8192 + 32768 + 65536);  // 8 MB
    float* bufB   = (float*)(ws + 8192 + 32768 + 65536 +
                             (size_t)GRAPHS * NODES * HID * 4);

    float* outf = (float*)d_out;

    int prep_n = 128 * 32 + 128 * 128 + 256 * 128;
    prep_weights<<<(prep_n + 255) / 256, 256, 0, stream>>>(W1, W2, Wc, W1T, W2T, WcT);
    mlp_in<<<GRAPHS, 256, 0, stream>>>(x_pf, W1T, b1, W2T, b2, bufA);
    edge_conv<<<GRAPHS, 256, 0, stream>>>(bufA, WcT, bc, bufB);  // f1
    edge_conv<<<GRAPHS, 256, 0, stream>>>(bufB, WcT, bc, bufA);  // f2
    edge_conv<<<GRAPHS, 256, 0, stream>>>(bufA, WcT, bc, bufB);  // f3
    head<<<GRAPHS, 128, 0, stream>>>(bufB, Wo1, bo1, Wo2, bo2, Wo3, bo3, Wo4, bo4, outf);
}
